// WeightedMultiheadAttention_39539468927021
// MI455X (gfx1250) — matrix operand order repaired and checked by reading, not compiled, not run on hardware
//
#include <hip/hip_runtime.h>

// ---------------- problem constants ----------------
#define TDIM 1024
#define EDIM 1024
#define BDIM 4
#define HDIM 16
#define DK   64
#define ROWS (BDIM * TDIM)   // 4096
#define SCALE 0.125f         // 1/sqrt(64)
#define WEPS  1e-5f

typedef __attribute__((ext_vector_type(16))) _Float16 v16h;
typedef __attribute__((ext_vector_type(8)))  _Float16 v8h;
typedef __attribute__((ext_vector_type(4)))  _Float16 v4h;
typedef __attribute__((ext_vector_type(8)))  float    v8f;

// ---------------- helpers ----------------
__device__ __forceinline__ v8f vzero() {
    v8f z;
#pragma unroll
    for (int i = 0; i < 8; ++i) z[i] = 0.0f;
    return z;
}

__device__ __forceinline__ v16h cat(v8h lo, v8h hi) {
    return __builtin_shufflevector(lo, hi, 0,1,2,3,4,5,6,7,8,9,10,11,12,13,14,15);
}

__device__ __forceinline__ v8f wmma16(v16h a, v16h b, v8f c) {
    // v_wmma_f32_16x16x32_f16 : D = A(16x32 f16) x B(32x16 f16) + C(f32)
    return __builtin_amdgcn_wmma_f32_16x16x32_f16(false, a, false, b, (short)0, c, false, false);
}

// A-fragment (16x32 MxK f16) from a row-major f16 matrix (lane: m=ln, kh=lane>>4)
__device__ __forceinline__ v16h load_a16(const _Float16* __restrict__ row, int kb, int khf) {
    v8h lo = *(const v8h*)(row + kb + khf * 8);
    v8h hi = *(const v8h*)(row + kb + 16 + khf * 8);
    return cat(lo, hi);
}

// B-fragment (32x16 KxN f16) from row n of the *transposed* factor (contiguous K)
__device__ __forceinline__ v16h load_b16(const _Float16* __restrict__ row, int kb, int khf) {
    v8h lo = *(const v8h*)(row + kb + khf * 8);
    v8h hi = *(const v8h*)(row + kb + 16 + khf * 8);
    return cat(lo, hi);
}

// =====================================================================
// Kernel 0: one-shot fp32 -> f16 conversion (elementwise, vectorized).
// =====================================================================
__global__ __launch_bounds__(256) void cvt16_kernel(const float* __restrict__ src,
                                                    _Float16* __restrict__ dst, int n4)
{
    const int i = blockIdx.x * blockDim.x + threadIdx.x;
    if (i >= n4) return;
    float4 x = ((const float4*)src)[i];
    v4h r;
    r[0] = (_Float16)x.x; r[1] = (_Float16)x.y;
    r[2] = (_Float16)x.z; r[3] = (_Float16)x.w;
    *(v4h*)(dst + (size_t)i * 4) = r;
}

// =====================================================================
// Software-pipelined 16x64 GEMM body: acc += Arow x Brows^T over EDIM.
// Double-buffers A + 4xB fragments so each WMMA waits on loads issued a
// full k-step earlier (partial loadcnt waits, ~20 loads in flight).
// =====================================================================
__device__ __forceinline__ void gemm_16x64_pipelined(
    const _Float16* __restrict__ arow, const _Float16* __restrict__ brow0,
    int khf, v8f acc[4])
{
    v16h aA, aB, bA[4], bB[4];

    aA = load_a16(arow, 0, khf);
#pragma unroll
    for (int j = 0; j < 4; ++j) bA[j] = load_b16(brow0 + (size_t)j * 16 * EDIM, 0, khf);

#pragma unroll 4
    for (int kb = 0; kb < EDIM; kb += 64) {
        // prefetch kb+32 while computing kb
        aB = load_a16(arow, kb + 32, khf);
#pragma unroll
        for (int j = 0; j < 4; ++j)
            bB[j] = load_b16(brow0 + (size_t)j * 16 * EDIM, kb + 32, khf);
#pragma unroll
        for (int j = 0; j < 4; ++j) acc[j] = wmma16(aA, bA[j], acc[j]);

        // prefetch kb+64 (branchless wrap; last prefetch is unused) while computing kb+32
        const int kn = (kb + 64) & (EDIM - 1);
        aA = load_a16(arow, kn, khf);
#pragma unroll
        for (int j = 0; j < 4; ++j)
            bA[j] = load_b16(brow0 + (size_t)j * 16 * EDIM, kn, khf);
#pragma unroll
        for (int j = 0; j < 4; ++j) acc[j] = wmma16(aB, bB[j], acc[j]);
    }
}

// =====================================================================
// Kernel 1: fused Q/K/V projections.
//   Xh = X @ W.T + b   (f16 operands, fp32 accum via WMMA)
//   Q,K stored [B*H, T, DK]; V stored transposed [B*H, DK, T] for P@V.
// n-tile-major decode: the 8 waves of a block fetch identical B frags.
// =====================================================================
__global__ __launch_bounds__(256) void proj_qkv_kernel(
    const _Float16* __restrict__ q16, const _Float16* __restrict__ k16,
    const _Float16* __restrict__ v16,
    const _Float16* __restrict__ Wq16, const _Float16* __restrict__ Wk16,
    const _Float16* __restrict__ Wv16,
    const float* __restrict__ bq, const float* __restrict__ bk,
    const float* __restrict__ bv,
    _Float16* __restrict__ qh, _Float16* __restrict__ khd, _Float16* __restrict__ vt)
{
    const int lane = threadIdx.x & 31;
    const int ln   = lane & 15;
    const int khf  = lane >> 4;
    const int wid  = blockIdx.x * (blockDim.x >> 5) + (threadIdx.x >> 5);

    const int MT = ROWS / 16;                 // 256 m-tiles
    const int NT = EDIM / 64;                 // 16 n-tiles of width 64
    const int mat   = wid / (MT * NT);
    const int rem   = wid % (MT * NT);
    const int ntile = rem / MT;               // n-tile-major: block shares B
    const int mtile = rem % MT;

    const _Float16* X    = (mat == 0) ? q16  : (mat == 1) ? k16  : v16;
    const _Float16* W    = (mat == 0) ? Wq16 : (mat == 1) ? Wk16 : Wv16;
    const float*    bias = (mat == 0) ? bq   : (mat == 1) ? bk   : bv;

    const _Float16* arow  = X + (size_t)(mtile * 16 + ln) * EDIM;
    const _Float16* brow0 = W + (size_t)(ntile * 64 + ln) * EDIM;

    v8f acc[4];
#pragma unroll
    for (int j = 0; j < 4; ++j) acc[j] = vzero();

    gemm_16x64_pipelined(arow, brow0, khf, acc);

#pragma unroll
    for (int j = 0; j < 4; ++j) {
        const int e  = ntile * 64 + j * 16 + ln;   // output column
        const float bval = bias[e];
        const int h = e >> 6, d = e & 63;
#pragma unroll
        for (int r = 0; r < 8; ++r) {
            const int gm   = mtile * 16 + khf * 8 + r;   // global row (b*T + t)
            const int bidx = gm >> 10;
            const int t    = gm & 1023;
            const _Float16 hv = (_Float16)(acc[j][r] + bval);
            if (mat == 0)      qh [((size_t)(bidx * HDIM + h) * TDIM + t) * DK + d] = hv;
            else if (mat == 1) khd[((size_t)(bidx * HDIM + h) * TDIM + t) * DK + d] = hv;
            else               vt [((size_t)(bidx * HDIM + h) * DK + d) * TDIM + t] = hv;
        }
    }
}

// =====================================================================
// Kernel 2: flash-style weighted attention per (b, h).
//   S = (Q Kh^T) * SCALE * w_k^2  (or -inf where w_k < eps)
//   online softmax over keys, O += P @ V  (V read from transposed layout)
// Block = 8 waves; wave owns 16 q-rows, k-tile = 64 keys/iteration.
// V fragments are loaded BEFORE the softmax VALU block so their latency
// hides under ~100 exp/shuffle/LDS instructions.
// =====================================================================
__global__ __launch_bounds__(256) void attn_kernel(
    const _Float16* __restrict__ qh, const _Float16* __restrict__ kh16,
    const _Float16* __restrict__ vt, const float* __restrict__ weights,
    _Float16* __restrict__ attn)
{
    __shared__ __align__(16) _Float16 Pt[8 * 16 * 64];   // per-wave 16x64 P tile (16 KB)

    const int lane = threadIdx.x & 31;
    const int ln   = lane & 15;
    const int khf  = lane >> 4;
    const int w    = threadIdx.x >> 5;

    const int qtiles = TDIM / 128;              // 8
    const int bh   = blockIdx.x / qtiles;       // b*16 + h
    const int qt   = blockIdx.x % qtiles;
    const int bidx = bh / HDIM;
    const int h    = bh % HDIM;
    const int qrow = qt * 128 + w * 16;

    // Q fragments for this wave's 16 rows (K-dim = DK = 64 -> 2 frags)
    const _Float16* Qrow = qh + ((size_t)bh * TDIM + qrow + ln) * DK;
    v16h qa[2];
#pragma unroll
    for (int s = 0; s < 2; ++s) qa[s] = load_a16(Qrow, s * 32, khf);

    v8f oacc[4];
#pragma unroll
    for (int j = 0; j < 4; ++j) oacc[j] = vzero();
    float mrow[8], lrow[8];
#pragma unroll
    for (int r = 0; r < 8; ++r) { mrow[r] = -3.0e38f; lrow[r] = 0.0f; }

    _Float16* Pw = &Pt[w * 16 * 64];

    for (int kt = 0; kt < TDIM; kt += 64) {
        // ---- batch-load all 8 K fragments, then 8 QK^T WMMAs ----
        v16h kfr[4][2];
#pragma unroll
        for (int j = 0; j < 4; ++j) {
            const _Float16* Krow = kh16 + ((size_t)bh * TDIM + kt + j * 16 + ln) * DK;
#pragma unroll
            for (int s = 0; s < 2; ++s) kfr[j][s] = load_b16(Krow, s * 32, khf);
        }
        v8f sacc[4];
#pragma unroll
        for (int j = 0; j < 4; ++j) sacc[j] = vzero();
#pragma unroll
        for (int j = 0; j < 4; ++j)
#pragma unroll
            for (int s = 0; s < 2; ++s)
                sacc[j] = wmma16(qa[s], kfr[j][s], sacc[j]);

        // ---- issue V fragment loads early: latency hides under softmax ----
        v16h vfr[4][2];
#pragma unroll
        for (int j = 0; j < 4; ++j) {
            const _Float16* Vrow = vt + ((size_t)bh * DK + j * 16 + ln) * TDIM + kt;
#pragma unroll
            for (int s = 0; s < 2; ++s) vfr[j][s] = load_b16(Vrow, s * 32, khf);
        }

        // ---- weighted mask + scale, row max ----
        float rmax[8];
#pragma unroll
        for (int r = 0; r < 8; ++r) rmax[r] = -3.0e38f;
#pragma unroll
        for (int j = 0; j < 4; ++j) {
            const float wv = weights[bidx * TDIM + kt + j * 16 + ln];
            const float f  = SCALE * wv * wv;
            const bool masked = wv < WEPS;
#pragma unroll
            for (int r = 0; r < 8; ++r) {
                float s = masked ? -3.0e38f : sacc[j][r] * f;
                sacc[j][r] = s;
                rmax[r] = fmaxf(rmax[r], s);
            }
        }
#pragma unroll
        for (int r = 0; r < 8; ++r) {
#pragma unroll
            for (int m = 1; m < 16; m <<= 1)
                rmax[r] = fmaxf(rmax[r], __shfl_xor(rmax[r], m, 32));
        }

        // ---- online softmax update ----
        float mnew[8], alpha[8], rsum[8];
#pragma unroll
        for (int r = 0; r < 8; ++r) {
            mnew[r]  = fmaxf(mrow[r], rmax[r]);
            alpha[r] = __expf(mrow[r] - mnew[r]);
            mrow[r]  = mnew[r];
            rsum[r]  = 0.0f;
        }
#pragma unroll
        for (int j = 0; j < 4; ++j) {
#pragma unroll
            for (int r = 0; r < 8; ++r) {
                float p = __expf(sacc[j][r] - mnew[r]);
                rsum[r] += p;
                Pw[(khf * 8 + r) * 64 + j * 16 + ln] = (_Float16)p;   // C-layout -> row-major
            }
        }
#pragma unroll
        for (int r = 0; r < 8; ++r) {
#pragma unroll
            for (int m = 1; m < 16; m <<= 1)
                rsum[r] += __shfl_xor(rsum[r], m, 32);
            lrow[r] = lrow[r] * alpha[r] + rsum[r];
        }
#pragma unroll
        for (int j = 0; j < 4; ++j)
#pragma unroll
            for (int r = 0; r < 8; ++r) oacc[j][r] *= alpha[r];

        // ---- O += P @ V (P re-fragmented via per-wave LDS; V already resident) ----
        v16h pa[2];
#pragma unroll
        for (int s = 0; s < 2; ++s) pa[s] = load_a16(Pw + ln * 64, s * 32, khf);
#pragma unroll
        for (int j = 0; j < 4; ++j)
#pragma unroll
            for (int s = 0; s < 2; ++s)
                oacc[j] = wmma16(pa[s], vfr[j][s], oacc[j]);
    }

    // ---- normalize and store attention output f16 [B, T, E] (heads re-packed) ----
#pragma unroll
    for (int j = 0; j < 4; ++j) {
        const int d = j * 16 + ln;
#pragma unroll
        for (int r = 0; r < 8; ++r) {
            const int t = qrow + khf * 8 + r;
            const float o = oacc[j][r] / lrow[r];
            attn[((size_t)bidx * TDIM + t) * EDIM + h * DK + d] = (_Float16)o;
        }
    }
}

// =====================================================================
// Kernel 3: output projection  out = Attn(f16) @ Wo.T + bo  (fp32 out)
// =====================================================================
__global__ __launch_bounds__(256) void out_proj_kernel(
    const _Float16* __restrict__ attn, const _Float16* __restrict__ Wo16,
    const float* __restrict__ bo, float* __restrict__ out)
{
    const int lane = threadIdx.x & 31;
    const int ln   = lane & 15;
    const int khf  = lane >> 4;
    const int wid  = blockIdx.x * (blockDim.x >> 5) + (threadIdx.x >> 5);

    const int MT = ROWS / 16;        // 256
    const int ntile = wid / MT;      // n-tile-major: block shares B
    const int mtile = wid % MT;

    const _Float16* arow  = attn + (size_t)(mtile * 16 + ln) * EDIM;
    const _Float16* brow0 = Wo16 + (size_t)(ntile * 64 + ln) * EDIM;

    v8f acc[4];
#pragma unroll
    for (int j = 0; j < 4; ++j) acc[j] = vzero();

    gemm_16x64_pipelined(arow, brow0, khf, acc);

#pragma unroll
    for (int j = 0; j < 4; ++j) {
        const int n = ntile * 64 + j * 16 + ln;
        const float bval = bo[n];
#pragma unroll
        for (int r = 0; r < 8; ++r) {
            const int gm = mtile * 16 + khf * 8 + r;
            out[(size_t)gm * EDIM + n] = acc[j][r] + bval;
        }
    }
}

// =====================================================================
extern "C" void kernel_launch(void* const* d_in, const int* in_sizes, int n_in,
                              void* d_out, int out_size, void* d_ws, size_t ws_size,
                              hipStream_t stream)
{
    const float* q       = (const float*)d_in[0];
    const float* k       = (const float*)d_in[1];
    const float* v       = (const float*)d_in[2];
    const float* weights = (const float*)d_in[3];
    const float* Wq      = (const float*)d_in[4];
    const float* bq      = (const float*)d_in[5];
    const float* Wk      = (const float*)d_in[6];
    const float* bk      = (const float*)d_in[7];
    const float* Wv      = (const float*)d_in[8];
    const float* bv      = (const float*)d_in[9];
    const float* Wo      = (const float*)d_in[10];
    const float* bo      = (const float*)d_in[11];

    // ---- workspace layout (halves): 64 MB total ----
    const size_t HS = (size_t)BDIM * HDIM * TDIM * DK;   // 4,194,304 head-buffer halves
    const size_t XS = (size_t)ROWS * EDIM;               // 4,194,304 input halves
    const size_t WS = (size_t)EDIM * EDIM;               // 1,048,576 weight halves
    _Float16* qh   = (_Float16*)d_ws;
    _Float16* kh16 = qh + HS;
    _Float16* vt   = kh16 + HS;
    _Float16* attn = vt + HS;
    _Float16* q16  = attn + HS;
    _Float16* k16  = q16 + XS;
    _Float16* v16  = k16 + XS;
    _Float16* wq16 = v16 + XS;
    _Float16* wk16 = wq16 + WS;
    _Float16* wv16 = wk16 + WS;
    _Float16* wo16 = wv16 + WS;

    // ---- stage 0: one-shot f32 -> f16 conversions ----
    const int XB = (int)(XS / 4 / 256);   // 4096 blocks
    const int WB = (int)(WS / 4 / 256);   // 1024 blocks
    cvt16_kernel<<<XB, 256, 0, stream>>>(q,  q16,  (int)(XS / 4));
    cvt16_kernel<<<XB, 256, 0, stream>>>(k,  k16,  (int)(XS / 4));
    cvt16_kernel<<<XB, 256, 0, stream>>>(v,  v16,  (int)(XS / 4));
    cvt16_kernel<<<WB, 256, 0, stream>>>(Wq, wq16, (int)(WS / 4));
    cvt16_kernel<<<WB, 256, 0, stream>>>(Wk, wk16, (int)(WS / 4));
    cvt16_kernel<<<WB, 256, 0, stream>>>(Wv, wv16, (int)(WS / 4));
    cvt16_kernel<<<WB, 256, 0, stream>>>(Wo, wo16, (int)(WS / 4));

    // ---- stage 1: Q/K/V projections (12288 waves / 8 per block) ----
    proj_qkv_kernel<<<1536, 256, 0, stream>>>(q16, k16, v16, wq16, wk16, wv16,
                                              bq, bk, bv, qh, kh16, vt);
    // ---- stage 2: attention (B*H*(T/128) = 512 blocks) ----
    attn_kernel<<<512, 256, 0, stream>>>(qh, kh16, vt, weights, attn);
    // ---- stage 3: output projection (4096 waves / 8 per block) ----
    out_proj_kernel<<<512, 256, 0, stream>>>(attn, wo16, bo, (float*)d_out);
}
